// Attention_2_branches_11038065951354
// MI455X (gfx1250) — matrix operand order repaired and checked by reading, not compiled, not run on hardware
//
#include <hip/hip_runtime.h>
#include <hip/hip_bf16.h>

// ---------------------------------------------------------------------------
// Two-branch MHA on MI455X (gfx1250, wave32).
// Compute-bound (~180 GFLOP vs ~100MB HBM traffic) => all GEMMs through
// v_wmma_f32_16x16x32_bf16 (fp32 accum); intermediates bf16; K/V/weight tiles
// staged with GLOBAL_LOAD_ASYNC_TO_LDS when the toolchain exposes it.
// ---------------------------------------------------------------------------

#define DIM 768
#define HEADS 12
#define HEAD_DIM 64
#define BATCH 8
#define SEQ 1024
#define SCALE 0.125f              // HEAD_DIM^-0.5
#define MROWS (2 * BATCH * SEQ)   // 16384: x rows then x2 rows

typedef __attribute__((ext_vector_type(16))) __bf16 v16bf;
typedef __attribute__((ext_vector_type(8)))  float  v8f;
typedef int v4i_vs __attribute__((vector_size(16)));   // matches builtin pointee

#if defined(__gfx1250__) && __has_builtin(__builtin_amdgcn_global_load_async_to_lds_b128)
#define HAVE_ASYNC_LDS 1
#else
#define HAVE_ASYNC_LDS 0
#endif

// 16-byte global->LDS copy: async (ASYNCcnt, no VGPR round trip) when available.
__device__ __forceinline__ void cp16(const __bf16* g, __bf16* l) {
#if HAVE_ASYNC_LDS
    __builtin_amdgcn_global_load_async_to_lds_b128(
        (__attribute__((address_space(1))) v4i_vs*)(uintptr_t)g,
        (__attribute__((address_space(3))) v4i_vs*)(uintptr_t)l,
        0, 0);
#else
    *(uint4*)l = *(const uint4*)g;
#endif
}
__device__ __forceinline__ void async_wait() {
#if HAVE_ASYNC_LDS
    asm volatile("s_wait_asynccnt 0" ::: "memory");
#endif
}

// ---- WMMA fragment loaders (contiguous k order, same in A and B) ------------------
// A 16x32 bf16: lane m=lane&15; k = ((lane>>4)&1)*16 + i
__device__ __forceinline__ v16bf load_frag_a(const __bf16* base, int ld, int lane) {
    const int m  = lane & 15;
    const int kh = (lane >> 4) & 1;
    const __bf16* p = base + m * ld + kh * 16;
    v16bf a;
#pragma unroll
    for (int i = 0; i < 16; ++i) a[i] = p[i];
    return a;
}
// B 32x16 bf16 stored natural [k][n]: lane n=lane&15; k = ((lane>>4)&1)*16 + i
__device__ __forceinline__ v16bf load_frag_b(const __bf16* base, int ld, int lane) {
    const int n  = lane & 15;
    const int k0 = ((lane >> 4) & 1) * 16;
    v16bf b;
#pragma unroll
    for (int i = 0; i < 16; ++i) b[i] = base[(k0 + i) * ld + n];
    return b;
}
// B 32x16 supplied transposed, i.e. base[n][k] row-major (row per output col)
__device__ __forceinline__ v16bf load_frag_bt(const __bf16* base, int ld, int lane) {
    const int n  = lane & 15;
    const int k0 = ((lane >> 4) & 1) * 16;
    const __bf16* p = base + n * ld + k0;
    v16bf b;
#pragma unroll
    for (int i = 0; i < 16; ++i) b[i] = p[i];
    return b;
}

#define WMMA_BF16(a, b, c) \
    __builtin_amdgcn_wmma_f32_16x16x32_bf16(false, (a), false, (b), (short)0, (c), false, false)

// ---------------------------------------------------------------------------
__global__ void cvt_f32_to_bf16(const float* __restrict__ in, __bf16* __restrict__ out, int n) {
    int i = blockIdx.x * 256 + threadIdx.x;
    if (i < n) out[i] = (__bf16)in[i];
}

// ---------------------------------------------------------------------------
// QKV GEMM: rows 0..8191 from x, 8192..16383 from x2; cols = qkv_w rows (2304).
// Block 256 thr (8 waves): 64x128 tile; wave = 16x64 strip; K step 32.
// ---------------------------------------------------------------------------
__global__ void qkv_gemm(const float* __restrict__ x, const float* __restrict__ x2,
                         const __bf16* __restrict__ wbf, __bf16* __restrict__ qkv_ws) {
    __shared__ __bf16 As[64][40];
    __shared__ __bf16 Bs[128][40];
    const int tid  = threadIdx.x;
    const int lane = tid & 31;
    const int wave = tid >> 5;
    const int row0 = blockIdx.y * 64;
    const int col0 = blockIdx.x * 128;
    const int srci = row0 >> 13;
    const float* xs = (srci == 0) ? x : x2;
    const int rr0 = row0 & 8191;
    const int wrow = wave & 3, wcol = wave >> 2;

    v8f acc[4] = {};
    for (int k0 = 0; k0 < DIM; k0 += 32) {
        __syncthreads();
        {   // stage A: 64x32 fp32 -> bf16 (8 elems/thread)
            int r = tid >> 2, cb = (tid & 3) * 8;
            const float* p = xs + (size_t)(rr0 + r) * DIM + k0 + cb;
#pragma unroll
            for (int i = 0; i < 8; ++i) As[r][cb + i] = (__bf16)p[i];
        }
        {   // stage B: 128x32 bf16 weight rows, async 16B copies
            int r = tid >> 1, cb = (tid & 1) * 16;
            const __bf16* p = wbf + (size_t)(col0 + r) * DIM + k0 + cb;
            cp16(p,     &Bs[r][cb]);
            cp16(p + 8, &Bs[r][cb + 8]);
        }
        async_wait();
        __syncthreads();
        v16bf a = load_frag_a(&As[wrow * 16][0], 40, lane);
#pragma unroll
        for (int t = 0; t < 4; ++t) {
            v16bf b = load_frag_bt(&Bs[wcol * 64 + t * 16][0], 40, lane);
            acc[t] = WMMA_BF16(a, b, acc[t]);
        }
    }
    // epilogue: scatter into [src,qkv,b,h,n,d] bf16
    const int half = lane >> 4, nn = lane & 15;
#pragma unroll
    for (int t = 0; t < 4; ++t) {
        int c  = col0 + wcol * 64 + t * 16 + nn;
        int qk = c / DIM;
        int cc = c % DIM;
        int h  = cc >> 6, d = cc & 63;
#pragma unroll
        for (int j = 0; j < 8; ++j) {
            int r  = row0 + wrow * 16 + half * 8 + j;
            int sr = r >> 13, rr = r & 8191;
            int b_ = rr >> 10, n_ = rr & 1023;
            size_t idx = ((((size_t)(sr * 3 + qk) * BATCH + b_) * HEADS + h) * SEQ + n_) * HEAD_DIM + d;
            qkv_ws[idx] = (__bf16)acc[t][j];
        }
    }
}

// ---------------------------------------------------------------------------
// Flash attention: block = 128 thr (4 waves) handles 64 q rows of one (b,h,br).
// 64-key steps: S = Q K^T (8 wmma), online softmax (shfl_xor row reduce),
// P via wave-private LDS into A-layout, O += P V (8 wmma).
// ---------------------------------------------------------------------------
__global__ void attn_kernel(const __bf16* __restrict__ qkv, __bf16* __restrict__ o_ws) {
    __shared__ __bf16 Qs[64][72];
    __shared__ __bf16 Ks[64][72];
    __shared__ __bf16 Vs[64][72];
    __shared__ __bf16 Ps[4][16][72];

    const int q0 = blockIdx.x * 64;
    const int bh = blockIdx.y;      // b*12 + h
    const int br = blockIdx.z;      // branch
    const int tid = threadIdx.x;
    const int lane = tid & 31;
    const int wave = tid >> 5;

    const __bf16* Qp = qkv + ((size_t)bh << 16);                          // src0,qkv0
    const __bf16* Kp = qkv + ((size_t)((br * 3 + 1) * 96 + bh) << 16);    // src=br,qkv1
    const __bf16* Vp = qkv + ((size_t)((br * 3 + 2) * 96 + bh) << 16);    // src=br,qkv2

    // stage Q tile (64x64 bf16) via async 16B copies
    for (int id = tid; id < 512; id += 128) {
        int r = id >> 3, c = (id & 7) * 8;
        cp16(Qp + (((size_t)(q0 + r)) << 6) + c, &Qs[r][c]);
    }

    float m_run[8], l_run[8];
#pragma unroll
    for (int j = 0; j < 8; ++j) { m_run[j] = -1.0e30f; l_run[j] = 0.0f; }
    v8f o[4] = {};

    for (int kb = 0; kb < SEQ; kb += 64) {
        __syncthreads();   // Ks/Vs free to overwrite
        for (int id = tid; id < 512; id += 128) {
            int r = id >> 3, c = (id & 7) * 8;
            cp16(Kp + (((size_t)(kb + r)) << 6) + c, &Ks[r][c]);
            cp16(Vp + (((size_t)(kb + r)) << 6) + c, &Vs[r][c]);
        }
        async_wait();
        __syncthreads();

        // prefetch next K/V tiles into cache while we compute
        if (kb + 64 < SEQ) {
            __builtin_prefetch(Kp + (((size_t)(kb + 64)) << 6) + ((size_t)tid << 6), 0, 3);
            __builtin_prefetch(Vp + (((size_t)(kb + 64)) << 6) + ((size_t)tid << 6), 0, 3);
        }

        // S = Q K^T : 16 q rows x 64 keys per wave
        v8f s[4] = {};
#pragma unroll
        for (int ds = 0; ds < 64; ds += 32) {
            v16bf aq = load_frag_a(&Qs[wave * 16][ds], 72, lane);
#pragma unroll
            for (int nt = 0; nt < 4; ++nt) {
                v16bf bk = load_frag_bt(&Ks[nt * 16][ds], 72, lane);
                s[nt] = WMMA_BF16(aq, bk, s[nt]);
            }
        }

        // online softmax per C-row (row m lives across 16 lanes of a half-wave)
        const int m_  = ((lane >> 4) << 3);
        const int nn_ = lane & 15;
#pragma unroll
        for (int j = 0; j < 8; ++j) {
            float v[4];
            float mx = -1.0e30f;
#pragma unroll
            for (int nt = 0; nt < 4; ++nt) { v[nt] = s[nt][j] * SCALE; mx = fmaxf(mx, v[nt]); }
            mx = fmaxf(mx, __shfl_xor(mx, 1, 32));
            mx = fmaxf(mx, __shfl_xor(mx, 2, 32));
            mx = fmaxf(mx, __shfl_xor(mx, 4, 32));
            mx = fmaxf(mx, __shfl_xor(mx, 8, 32));
            float mnew  = fmaxf(m_run[j], mx);
            float alpha = __expf(m_run[j] - mnew);
            m_run[j] = mnew;
            float ps = 0.0f;
#pragma unroll
            for (int nt = 0; nt < 4; ++nt) {
                float p = __expf(v[nt] - mnew);
                ps += p;
                Ps[wave][m_ + j][nt * 16 + nn_] = (__bf16)p;
            }
            ps += __shfl_xor(ps, 1, 32);
            ps += __shfl_xor(ps, 2, 32);
            ps += __shfl_xor(ps, 4, 32);
            ps += __shfl_xor(ps, 8, 32);
            l_run[j] = l_run[j] * alpha + ps;
#pragma unroll
            for (int t = 0; t < 4; ++t) o[t][j] *= alpha;
        }
        // Ps store->load is same-wave LDS (in-order; compiler inserts dscnt wait)

        // O += P V  (two 32-key chunks)
#pragma unroll
        for (int kc = 0; kc < 64; kc += 32) {
            v16bf ap = load_frag_a(&Ps[wave][0][kc], 72, lane);
#pragma unroll
            for (int t = 0; t < 4; ++t) {
                v16bf bv = load_frag_b(&Vs[kc][t * 16], 72, lane);
                o[t] = WMMA_BF16(ap, bv, o[t]);
            }
        }
    }

    // epilogue: O /= l ; store bf16 [br, b, n, h*64+d]
    const int b_ = bh / HEADS, h_ = bh % HEADS;
    const int half = lane >> 4, nn = lane & 15;
#pragma unroll
    for (int j = 0; j < 8; ++j) {
        float invl = 1.0f / l_run[j];
        int n_ = q0 + wave * 16 + half * 8 + j;
        size_t rb = ((size_t)(br * BATCH + b_) * SEQ + n_) * DIM + h_ * HEAD_DIM;
#pragma unroll
        for (int t = 0; t < 4; ++t)
            o_ws[rb + t * 16 + nn] = (__bf16)(o[t][j] * invl);
    }
}

// ---------------------------------------------------------------------------
// Output projection: out = O @ proj_w^T + b, M=16384 (branch0 rows ‖ branch1),
// matching d_out = out1 ‖ out2. fp32 output.
// ---------------------------------------------------------------------------
__global__ void proj_gemm(const __bf16* __restrict__ obf, const __bf16* __restrict__ wbf,
                          const float* __restrict__ bias, float* __restrict__ out) {
    __shared__ __bf16 As[64][40];
    __shared__ __bf16 Bs[128][40];
    const int tid  = threadIdx.x;
    const int lane = tid & 31;
    const int wave = tid >> 5;
    const int row0 = blockIdx.y * 64;
    const int col0 = blockIdx.x * 128;
    const int wrow = wave & 3, wcol = wave >> 2;

    v8f acc[4] = {};
    for (int k0 = 0; k0 < DIM; k0 += 32) {
        __syncthreads();
        {   // stage A: 64x32 bf16, async
            int r = tid >> 2, cb = (tid & 3) * 8;
            cp16(obf + (size_t)(row0 + r) * DIM + k0 + cb, &As[r][cb]);
        }
        {   // stage B: 128x32 bf16, async
            int r = tid >> 1, cb = (tid & 1) * 16;
            const __bf16* p = wbf + (size_t)(col0 + r) * DIM + k0 + cb;
            cp16(p,     &Bs[r][cb]);
            cp16(p + 8, &Bs[r][cb + 8]);
        }
        async_wait();
        __syncthreads();
        v16bf a = load_frag_a(&As[wrow * 16][0], 40, lane);
#pragma unroll
        for (int t = 0; t < 4; ++t) {
            v16bf b = load_frag_bt(&Bs[wcol * 64 + t * 16][0], 40, lane);
            acc[t] = WMMA_BF16(a, b, acc[t]);
        }
    }
    const int half = lane >> 4, nn = lane & 15;
#pragma unroll
    for (int t = 0; t < 4; ++t) {
        int c = col0 + wcol * 64 + t * 16 + nn;
        float bv = bias[c];
#pragma unroll
        for (int j = 0; j < 8; ++j) {
            int r = row0 + wrow * 16 + half * 8 + j;
            out[(size_t)r * DIM + c] = acc[t][j] + bv;
        }
    }
}

// ---------------------------------------------------------------------------
extern "C" void kernel_launch(void* const* d_in, const int* in_sizes, int n_in,
                              void* d_out, int out_size, void* d_ws, size_t ws_size,
                              hipStream_t stream) {
    const float* x      = (const float*)d_in[0];
    const float* x2     = (const float*)d_in[1];
    const float* qkv_w  = (const float*)d_in[2];
    const float* proj_w = (const float*)d_in[3];
    const float* proj_b = (const float*)d_in[4];
    float* out = (float*)d_out;

    // workspace carve-up (bf16 elements)
    __bf16* qkvw_bf  = (__bf16*)d_ws;                      // 3*768*768
    __bf16* projw_bf = qkvw_bf + 3 * DIM * DIM;            // 768*768
    __bf16* qkv_ws   = projw_bf + DIM * DIM;               // 2*3*8192*768
    __bf16* o_ws     = qkv_ws + (size_t)2 * 3 * BATCH * SEQ * DIM;  // 2*8192*768

    {
        int n1 = 3 * DIM * DIM;
        cvt_f32_to_bf16<<<(n1 + 255) / 256, 256, 0, stream>>>(qkv_w, qkvw_bf, n1);
        int n2 = DIM * DIM;
        cvt_f32_to_bf16<<<(n2 + 255) / 256, 256, 0, stream>>>(proj_w, projw_bf, n2);
    }
    {
        dim3 grid(3 * DIM / 128, MROWS / 64);   // (18, 256)
        qkv_gemm<<<grid, 256, 0, stream>>>(x, x2, qkvw_bf, qkv_ws);
    }
    {
        dim3 grid(SEQ / 64, BATCH * HEADS, 2);  // (16, 96, 2)
        attn_kernel<<<grid, 128, 0, stream>>>(qkv_ws, o_ws);
    }
    {
        dim3 grid(DIM / 128, MROWS / 64);       // (6, 256)
        proj_gemm<<<grid, 256, 0, stream>>>(o_ws, projw_bf, proj_b, out);
    }
}
